// HandModel_3925600108909
// MI455X (gfx1250) — matrix operand order repaired and checked by reading, not compiled, not run on hardware
//
#include <hip/hip_runtime.h>
#include <hip/hip_bf16.h>
#include <stdint.h>

typedef __attribute__((ext_vector_type(8)))  __bf16 v8bf;
typedef __attribute__((ext_vector_type(16))) __bf16 v16bf;
typedef __attribute__((ext_vector_type(8)))  float  v8f;

#define NB     32
#define N1P    21
#define N2P    512
#define KNN    64
#define NCBIG  (NB*KNN*N1P)   /* 43008 big-column count (b,k,n1), n1 fastest */
#define KN1    (KNN*N1P)      /* 1344 */
#define NS     672            /* B*N1 small columns */
#define SSTR   704            /* small columns padded to 16*44 = 704 */

__device__ __forceinline__ float lrelu(float x) { return x > 0.f ? x : 0.1f * x; }

// ---------------- fp32 -> bf16 weight convert with K-padding ----------------
__global__ void conv_w_kernel(const float* __restrict__ w, __bf16* __restrict__ wb,
                              int O, int Cin, int Cpad) {
  int i = blockIdx.x * 256 + threadIdx.x;
  if (i >= O * Cpad) return;
  int o = i / Cpad, c = i - o * Cpad;
  float v = (c < Cin) ? w[(size_t)o * Cin + c] : 0.f;
  wb[i] = (__bf16)v;
}

// ---------------- KNN: block per (b,n1), iterative 64x arg-min -------------
__global__ void knn_kernel(const float* __restrict__ xyz1, const float* __restrict__ xyz2,
                           int* __restrict__ idx) {
  int bn = blockIdx.x;                  // b*21 + n1
  int b = bn / N1P, n1 = bn - b * N1P;
  __shared__ float dist[N2P];
  __shared__ unsigned long long red[256];
  int t = threadIdx.x;
  float px = xyz1[(b * 3 + 0) * N1P + n1];
  float py = xyz1[(b * 3 + 1) * N1P + n1];
  float pz = xyz1[(b * 3 + 2) * N1P + n1];
  for (int j = t; j < N2P; j += 256) {
    float dx = xyz2[(b * 3 + 0) * N2P + j] - px;
    float dy = xyz2[(b * 3 + 1) * N2P + j] - py;
    float dz = xyz2[(b * 3 + 2) * N2P + j] - pz;
    dist[j] = dx * dx + dy * dy + dz * dz;
  }
  __syncthreads();
  for (int s = 0; s < KNN; ++s) {
    unsigned long long best = ~0ull;
    for (int j = t; j < N2P; j += 256) {
      unsigned u = __float_as_uint(dist[j]);
      u = (u & 0x80000000u) ? ~u : (u | 0x80000000u);  // ordered-uint transform
      unsigned long long key = ((unsigned long long)u << 32) | (unsigned)j;
      if (key < best) best = key;
    }
    red[t] = best;
    __syncthreads();
    for (int off = 128; off > 0; off >>= 1) {
      if (t < off) { unsigned long long o2 = red[t + off]; if (o2 < red[t]) red[t] = o2; }
      __syncthreads();
    }
    int jw = (int)(red[0] & 0xffffffffu);
    if (t == 0) {
      idx[bn * KNN + s] = jw;
      dist[jw] = __int_as_float(0x7f800000);  // +inf: remove from later rounds
    }
    __syncthreads();
  }
}

// ------------- build new_points as bf16 [288][43008] (gather + concat) ------
__global__ void build_xin(const float* __restrict__ points2, const float* __restrict__ xyz1,
                          const float* __restrict__ xyz2, const int* __restrict__ idx,
                          __bf16* __restrict__ xin) {
  int col = blockIdx.x * 256 + threadIdx.x;
  int c = blockIdx.y;
  if (col >= NCBIG) return;
  int b = col / KN1, rem = col - b * KN1;
  int k = rem / N1P, n1 = rem - k * N1P;
  int j = idx[(b * N1P + n1) * KNN + k];
  float v;
  if (c < 256)      v = points2[((size_t)b * 256 + c) * N2P + j];
  else if (c < 259) { int d = c - 256;
                      v = xyz2[((size_t)b * 3 + d) * N2P + j] - xyz1[((size_t)b * 3 + d) * N1P + n1]; }
  else              v = 0.f;  // K-dim zero padding 259..287
  xin[(size_t)c * NCBIG + col] = (__bf16)v;
}

// -------- per-channel graph mixing: tmp[c][b*21+m] = sum_n p1[b,c,n]*A[c,n,m] ----
__global__ void build_tmpx(const float* __restrict__ points1, const float* __restrict__ ga,
                           __bf16* __restrict__ tmpx) {
  int sp = blockIdx.x * 256 + threadIdx.x;
  int c = blockIdx.y;
  if (sp >= SSTR) return;
  float acc = 0.f;
  if (sp < NS) {
    int b = sp / N1P, m = sp - b * N1P;
    const float* p1 = points1 + ((size_t)b * 512 + c) * N1P;
    const float* aa = ga + (size_t)c * (N1P * N1P) + m;
    #pragma unroll
    for (int n = 0; n < N1P; ++n) acc += p1[n] * aa[n * N1P];
  }
  tmpx[(size_t)c * SSTR + sp] = (__bf16)acc;  // pad cols -> 0
}

// ---------------- reductions over the K (neighbor) axis --------------------
__global__ void reduce_max_k(const __bf16* __restrict__ in, __bf16* __restrict__ out) {
  int sp = blockIdx.x * 256 + threadIdx.x;
  int c = blockIdx.y;
  if (sp >= SSTR) return;
  float m = 0.f;
  if (sp < NS) {
    int b = sp / N1P, n1 = sp - b * N1P;
    const __bf16* p = in + (size_t)c * NCBIG + (size_t)b * KN1 + n1;
    m = -3.0e38f;
    #pragma unroll 4
    for (int k = 0; k < KNN; ++k) m = fmaxf(m, (float)p[k * N1P]);
  }
  out[(size_t)c * SSTR + sp] = (__bf16)m;
}

__global__ void reduce_mean_k(const __bf16* __restrict__ in, float* __restrict__ out) {
  int sp = blockIdx.x * 256 + threadIdx.x;
  int c = blockIdx.y;
  if (sp >= SSTR) return;
  float s = 0.f;
  if (sp < NS) {
    int b = sp / N1P, n1 = sp - b * N1P;
    const __bf16* p = in + (size_t)c * NCBIG + (size_t)b * KN1 + n1;
    #pragma unroll 4
    for (int k = 0; k < KNN; ++k) s += (float)p[k * N1P];
    s *= (1.f / (float)KNN);
  }
  out[(size_t)c * SSTR + sp] = s;
}

// =============================================================================
// WMMA GEMM  D[o][col] = sum_c W[o][c] * X[c][col]  with fused epilogues.
// Block tile 128x64, 8 wave32s, each wave 32x32 = 2x2 v_wmma_f32_16x16x32_bf16.
// MODE: 0 pg(leaky+bias, f32+bf16 out)  1 plain f32 out   2 leaky(bias+bcast)
//       3 leaky(bias)  4 sigmoid-gate -> g1 / h=g2*pg  5 bias -> f32
//       6 plain bf16   7 leaky(acc*g1+pe)+points1      8 leaky((acc+b)*g1+pe)
//       9 leaky(bias)+vres -> d_out
// =============================================================================
template <int MODE>
__global__ __launch_bounds__(256)
void gemm_ep(const __bf16* __restrict__ W, const __bf16* __restrict__ X,
             int O, int Cpad, int NCOL,
             const float* __restrict__ bias,
             const float* __restrict__ auxf,          // fqpg/fkpg/k2c/vres [.][704]
             const float* __restrict__ auxp,          // pg f32 (mode4) / points1 (mode7)
             const __bf16* __restrict__ g1p, const __bf16* __restrict__ pep,
             __bf16* __restrict__ outb, __bf16* __restrict__ outb2,
             float* __restrict__ outf) {
  __shared__ __align__(16) __bf16 Ws[128 * 40];  // rows padded 32->40 (80B stride)
  __shared__ __align__(16) __bf16 Xs[64 * 40];   // transposed: [col][k]
  const int ro = blockIdx.y * 128;
  const int co = blockIdx.x * 64;
  const int t = threadIdx.x;
  const int lane = t & 31, wid = t >> 5;
  const int wm = (wid & 3) * 32;   // wave row offset in block tile
  const int wn = (wid >> 2) * 32;  // wave col offset
  const int half = lane >> 4, l15 = lane & 15;
  const int akb = half * 16;       // A-frag K base: contiguous K per lane half

  const int wrow = t >> 1, wcol = (t & 1) * 16;  // W-tile load mapping
  const int xk = t >> 3, xc = (t & 7) * 8;       // X-tile load mapping

  v8f acc[2][2] = {};

  for (int kk = 0; kk < Cpad; kk += 32) {
    {  // W tile 128x32, two b128 loads per thread
      const __bf16* gp = W + (size_t)(ro + wrow) * Cpad + (kk + wcol);
      *(v8bf*)&Ws[wrow * 40 + wcol] = *(const v8bf*)gp;
      *(v8bf*)&Ws[wrow * 40 + wcol + 8] = *(const v8bf*)(gp + 8);
    }
    {  // X tile 32x64, coalesced b128 load, transposed store
      const __bf16* gp = X + (size_t)(kk + xk) * NCOL + (co + xc);
      v8bf xv = *(const v8bf*)gp;
      #pragma unroll
      for (int j = 0; j < 8; ++j) Xs[(xc + j) * 40 + xk] = xv[j];
    }
    __syncthreads();

    v16bf a0, a1, b0, b1;
    {  // A frags: lane half selects contiguous K 0-15 vs 16-31
      const __bf16* p = &Ws[(wm + l15) * 40 + akb];
      v8bf lo = *(const v8bf*)p, hi = *(const v8bf*)(p + 8);
      a0 = __builtin_shufflevector(lo, hi, 0,1,2,3,4,5,6,7,8,9,10,11,12,13,14,15);
    }
    {
      const __bf16* p = &Ws[(wm + 16 + l15) * 40 + akb];
      v8bf lo = *(const v8bf*)p, hi = *(const v8bf*)(p + 8);
      a1 = __builtin_shufflevector(lo, hi, 0,1,2,3,4,5,6,7,8,9,10,11,12,13,14,15);
    }
    {  // B frags: lane half selects contiguous K 0-15 vs 16-31
      const __bf16* p = &Xs[(wn + l15) * 40 + half * 16];
      v8bf lo = *(const v8bf*)p, hi = *(const v8bf*)(p + 8);
      b0 = __builtin_shufflevector(lo, hi, 0,1,2,3,4,5,6,7,8,9,10,11,12,13,14,15);
    }
    {
      const __bf16* p = &Xs[(wn + 16 + l15) * 40 + half * 16];
      v8bf lo = *(const v8bf*)p, hi = *(const v8bf*)(p + 8);
      b1 = __builtin_shufflevector(lo, hi, 0,1,2,3,4,5,6,7,8,9,10,11,12,13,14,15);
    }
    acc[0][0] = __builtin_amdgcn_wmma_f32_16x16x32_bf16(false, a0, false, b0, (short)0, acc[0][0], false, false);
    acc[0][1] = __builtin_amdgcn_wmma_f32_16x16x32_bf16(false, a0, false, b1, (short)0, acc[0][1], false, false);
    acc[1][0] = __builtin_amdgcn_wmma_f32_16x16x32_bf16(false, a1, false, b0, (short)0, acc[1][0], false, false);
    acc[1][1] = __builtin_amdgcn_wmma_f32_16x16x32_bf16(false, a1, false, b1, (short)0, acc[1][1], false, false);
    __syncthreads();
  }

  // epilogue: C/D layout -> row = half*8 + j, col = l15 within each 16x16 tile
  #pragma unroll
  for (int mt = 0; mt < 2; ++mt) {
    #pragma unroll
    for (int nt = 0; nt < 2; ++nt) {
      #pragma unroll
      for (int j = 0; j < 8; ++j) {
        const int o = ro + wm + mt * 16 + half * 8 + j;
        const int col = co + wn + nt * 16 + l15;
        float v = acc[mt][nt][j];
        if (MODE == 0) {
          v = lrelu(v + bias[o]);
          outf[(size_t)o * SSTR + col] = v;
          outb[(size_t)o * SSTR + col] = (__bf16)v;
        } else if (MODE == 1) {
          outf[(size_t)o * SSTR + col] = v;
        } else if (MODE == 2) {
          int b = col / KN1, rem = col - b * KN1;
          int scol = b * N1P + (rem % N1P);
          v = lrelu(v + bias[o] + auxf[(size_t)o * SSTR + scol]);
          outb[(size_t)o * NCOL + col] = (__bf16)v;
        } else if (MODE == 3) {
          v = lrelu(v + bias[o]);
          outb[(size_t)o * NCOL + col] = (__bf16)v;
        } else if (MODE == 4) {
          int b = col / KN1, rem = col - b * KN1;
          int scol = b * N1P + (rem % N1P);
          float s = 1.f / (1.f + __expf(-((v + bias[o]) * auxf[(size_t)o * SSTR + scol])));
          if (o < 512) outb[(size_t)o * NCOL + col] = (__bf16)s;               // g1
          else outb2[(size_t)(o - 512) * NCOL + col] =
                   (__bf16)(s * auxp[(size_t)(o - 512) * SSTR + scol]);        // h = g2*pg
        } else if (MODE == 5) {
          outf[(size_t)o * SSTR + col] = v + bias[o];
        } else if (MODE == 6) {
          outb[(size_t)o * NCOL + col] = (__bf16)v;
        } else if (MODE == 7) {
          int b = col / KN1, rem = col - b * KN1;
          int n1 = rem % N1P;
          float g = (float)g1p[(size_t)o * NCOL + col];
          float p = (float)pep[(size_t)o * NCOL + col];
          v = lrelu(v * g + p) + auxp[((size_t)b * 512 + o) * N1P + n1];
          outb[(size_t)o * NCOL + col] = (__bf16)v;
        } else if (MODE == 8) {
          float g = (float)g1p[(size_t)o * NCOL + col];
          float p = (float)pep[(size_t)o * NCOL + col];
          v = lrelu((v + bias[o]) * g + p);
          outb[(size_t)o * NCOL + col] = (__bf16)v;
        } else if (MODE == 9) {
          if (col < NS) {
            int b = col / N1P, n1 = col - b * N1P;
            outf[((size_t)b * 512 + o) * N1P + n1] =
                lrelu(v + bias[o]) + auxf[(size_t)o * SSTR + col];
          }
        }
      }
    }
  }
}

extern "C" void kernel_launch(void* const* d_in, const int* in_sizes, int n_in,
                              void* d_out, int out_size, void* d_ws, size_t ws_size,
                              hipStream_t stream) {
  (void)in_sizes; (void)n_in; (void)out_size; (void)ws_size;
  const float* xyz1 = (const float*)d_in[0];
  const float* xyz2 = (const float*)d_in[1];
  const float* points1 = (const float*)d_in[2];
  const float* points2 = (const float*)d_in[3];
  const float* graph_a = (const float*)d_in[4];
  const float* gw_w = (const float*)d_in[5];  const float* gw_b = (const float*)d_in[6];
  const float* fq_w = (const float*)d_in[7];
  const float* fk_w = (const float*)d_in[8];
  const float* fv1_w = (const float*)d_in[9];
  const float* fv2_w = (const float*)d_in[10];
  const float* q0_w = (const float*)d_in[11]; const float* q0_b = (const float*)d_in[12];
  const float* q1_w = (const float*)d_in[13]; const float* q1_b = (const float*)d_in[14];
  const float* q2_w = (const float*)d_in[15]; const float* q2_b = (const float*)d_in[16];
  const float* k0_w = (const float*)d_in[17]; const float* k0_b = (const float*)d_in[18];
  const float* k1_w = (const float*)d_in[19]; const float* k1_b = (const float*)d_in[20];
  const float* k2_w = (const float*)d_in[21]; const float* k2_b = (const float*)d_in[22];
  const float* v0_w = (const float*)d_in[23]; const float* v0_b = (const float*)d_in[24];
  const float* v1_w = (const float*)d_in[25]; const float* v1_b = (const float*)d_in[26];
  const float* v2_w = (const float*)d_in[27]; const float* v2_b = (const float*)d_in[28];

  char* ws = (char*)d_ws;
  size_t off = 0;
  auto alloc = [&](size_t bytes) -> char* {
    char* p = ws + off;
    off += (bytes + 255) & ~(size_t)255;
    return p;
  };

  __bf16* wb_gw  = (__bf16*)alloc((size_t)512 * 512 * 2);
  __bf16* wb_fq  = (__bf16*)alloc((size_t)512 * 512 * 2);
  __bf16* wb_fk  = (__bf16*)alloc((size_t)512 * 512 * 2);
  __bf16* wb_fv1 = (__bf16*)alloc((size_t)512 * 288 * 2);
  __bf16* wb_fv2 = (__bf16*)alloc((size_t)512 * 512 * 2);
  __bf16* wb_q0  = (__bf16*)alloc((size_t)512 * 288 * 2);
  __bf16* wb_q1  = (__bf16*)alloc((size_t)512 * 512 * 2);
  __bf16* wb_q2  = (__bf16*)alloc((size_t)1024 * 512 * 2);
  __bf16* wb_k0  = (__bf16*)alloc((size_t)512 * 288 * 2);
  __bf16* wb_k1  = (__bf16*)alloc((size_t)512 * 512 * 2);
  __bf16* wb_k2  = (__bf16*)alloc((size_t)1024 * 512 * 2);
  __bf16* wb_v0  = (__bf16*)alloc((size_t)512 * 512 * 2);
  __bf16* wb_v1  = (__bf16*)alloc((size_t)512 * 512 * 2);
  __bf16* wb_v2  = (__bf16*)alloc((size_t)512 * 512 * 2);
  int*    idxb = (int*)alloc((size_t)NS * KNN * 4);
  __bf16* xin  = (__bf16*)alloc((size_t)288 * NCBIG * 2);
  __bf16* tmpx = (__bf16*)alloc((size_t)512 * SSTR * 2);
  float*  pgf  = (float*)alloc((size_t)512 * SSTR * 4);
  __bf16* pgb  = (__bf16*)alloc((size_t)512 * SSTR * 2);
  float*  fqpg = (float*)alloc((size_t)512 * SSTR * 4);
  float*  fkpg = (float*)alloc((size_t)512 * SSTR * 4);
  float*  k2c  = (float*)alloc((size_t)1024 * SSTR * 4);
  __bf16* kmax = (__bf16*)alloc((size_t)512 * SSTR * 2);
  __bf16* vmax = (__bf16*)alloc((size_t)512 * SSTR * 2);
  float*  vres = (float*)alloc((size_t)512 * SSTR * 4);
  __bf16* buf1 = (__bf16*)alloc((size_t)512 * NCBIG * 2);  // qa -> g1
  __bf16* buf2 = (__bf16*)alloc((size_t)512 * NCBIG * 2);  // ka -> pe
  __bf16* buf3 = (__bf16*)alloc((size_t)512 * NCBIG * 2);  // qb -> vcomb -> vb
  __bf16* buf4 = (__bf16*)alloc((size_t)512 * NCBIG * 2);  // kb -> h -> va

  auto cw = [&](const float* w, __bf16* wb, int O, int Cin, int Cpad) {
    int n = O * Cpad;
    conv_w_kernel<<<(n + 255) / 256, 256, 0, stream>>>(w, wb, O, Cin, Cpad);
  };
  cw(gw_w, wb_gw, 512, 512, 512);   cw(fq_w, wb_fq, 512, 512, 512);
  cw(fk_w, wb_fk, 512, 512, 512);   cw(fv1_w, wb_fv1, 512, 259, 288);
  cw(fv2_w, wb_fv2, 512, 512, 512); cw(q0_w, wb_q0, 512, 259, 288);
  cw(q1_w, wb_q1, 512, 512, 512);   cw(q2_w, wb_q2, 1024, 512, 512);
  cw(k0_w, wb_k0, 512, 259, 288);   cw(k1_w, wb_k1, 512, 512, 512);
  cw(k2_w, wb_k2, 1024, 512, 512);  cw(v0_w, wb_v0, 512, 512, 512);
  cw(v1_w, wb_v1, 512, 512, 512);   cw(v2_w, wb_v2, 512, 512, 512);

  knn_kernel<<<NS, 256, 0, stream>>>(xyz1, xyz2, idxb);
  build_xin<<<dim3(NCBIG / 256, 288), 256, 0, stream>>>(points2, xyz1, xyz2, idxb, xin);
  build_tmpx<<<dim3(3, 512), 256, 0, stream>>>(points1, graph_a, tmpx);

  const dim3 gS(SSTR / 64, 4), gS2(SSTR / 64, 8);   // small GEMMs (672->704 cols)
  const dim3 gB(NCBIG / 64, 4), gB2(NCBIG / 64, 8); // big GEMMs (43008 cols)

  // pg = leaky(gw @ tmp + b)  -> f32 + bf16
  gemm_ep<0><<<gS, 256, 0, stream>>>(wb_gw, tmpx, 512, 512, SSTR, gw_b,
      nullptr, nullptr, nullptr, nullptr, pgb, nullptr, pgf);
  // fqpg / fkpg = fq/fk @ pg
  gemm_ep<1><<<gS, 256, 0, stream>>>(wb_fq, pgb, 512, 512, SSTR, nullptr,
      nullptr, nullptr, nullptr, nullptr, nullptr, nullptr, fqpg);
  gemm_ep<1><<<gS, 256, 0, stream>>>(wb_fk, pgb, 512, 512, SSTR, nullptr,
      nullptr, nullptr, nullptr, nullptr, nullptr, nullptr, fkpg);
  // qa = leaky(q0 @ xin + b + fqpg)   ka = leaky(k0 @ xin + b + fkpg)
  gemm_ep<2><<<gB, 256, 0, stream>>>(wb_q0, xin, 512, 288, NCBIG, q0_b,
      fqpg, nullptr, nullptr, nullptr, buf1, nullptr, nullptr);
  gemm_ep<2><<<gB, 256, 0, stream>>>(wb_k0, xin, 512, 288, NCBIG, k0_b,
      fkpg, nullptr, nullptr, nullptr, buf2, nullptr, nullptr);
  // qb = leaky(q1 @ qa + b)           kb = leaky(k1 @ ka + b)
  gemm_ep<3><<<gB, 256, 0, stream>>>(wb_q1, buf1, 512, 512, NCBIG, q1_b,
      nullptr, nullptr, nullptr, nullptr, buf3, nullptr, nullptr);
  gemm_ep<3><<<gB, 256, 0, stream>>>(wb_k1, buf2, 512, 512, NCBIG, k1_b,
      nullptr, nullptr, nullptr, nullptr, buf4, nullptr, nullptr);
  // kmax = max_k kb ; k2c = k2 @ kmax + b
  reduce_max_k<<<dim3(3, 512), 256, 0, stream>>>(buf4, kmax);
  gemm_ep<5><<<gS2, 256, 0, stream>>>(wb_k2, kmax, 1024, 512, SSTR, k2_b,
      nullptr, nullptr, nullptr, nullptr, nullptr, nullptr, k2c);
  // a = sigmoid(k2c * (q2@qb + b)) -> g1 (buf1) and h = g2*pg (buf4)
  gemm_ep<4><<<gB2, 256, 0, stream>>>(wb_q2, buf3, 1024, 512, NCBIG, q2_b,
      k2c, pgf, nullptr, nullptr, buf1, buf4, nullptr);
  // pe = fv2 @ h
  gemm_ep<6><<<gB, 256, 0, stream>>>(wb_fv2, buf4, 512, 512, NCBIG, nullptr,
      nullptr, nullptr, nullptr, nullptr, buf2, nullptr, nullptr);
  // vcomb = leaky((fv1@xin)*g1 + pe) + points1
  gemm_ep<7><<<gB, 256, 0, stream>>>(wb_fv1, xin, 512, 288, NCBIG, nullptr,
      nullptr, points1, buf1, buf2, buf3, nullptr, nullptr);
  // v_res = mean_k vcomb
  reduce_mean_k<<<dim3(3, 512), 256, 0, stream>>>(buf3, vres);
  // va = leaky((v0@vcomb + b)*g1 + pe)
  gemm_ep<8><<<gB, 256, 0, stream>>>(wb_v0, buf3, 512, 512, NCBIG, v0_b,
      nullptr, nullptr, buf1, buf2, buf4, nullptr, nullptr);
  // vb = leaky(v1@va + b)
  gemm_ep<3><<<gB, 256, 0, stream>>>(wb_v1, buf4, 512, 512, NCBIG, v1_b,
      nullptr, nullptr, nullptr, nullptr, buf3, nullptr, nullptr);
  // vmax = max_k vb ; out = leaky(v2@vmax + b) + v_res
  reduce_max_k<<<dim3(3, 512), 256, 0, stream>>>(buf3, vmax);
  gemm_ep<9><<<gS, 256, 0, stream>>>(wb_v2, vmax, 512, 512, SSTR, v2_b,
      vres, nullptr, nullptr, nullptr, nullptr, nullptr, (float*)d_out);
}
